// LBPModule_28518582845782
// MI455X (gfx1250) — matrix operand order repaired and checked by reading, not compiled, not run on hardware
//
#include <hip/hip_runtime.h>

typedef __attribute__((ext_vector_type(16))) _Float16 v16h;
typedef __attribute__((ext_vector_type(8)))  _Float16 v8h;
typedef __attribute__((ext_vector_type(8)))  float    v8f;

// ---------------- problem dimensions ----------------
#define B_    4
#define F_    16
#define H_    224
#define W_    224
#define HW_   (H_*W_)          // 50176
#define C1_   64
#define H1_   112
#define W1_   112
#define HW1_  (H1_*W1_)        // 12544
#define C2_   128
#define F2_   8
#define H2_   56
#define W2_   56
#define HW2_  (H2_*W2_)        // 3136
#define N1_   (B_*F_*HW1_)     // 802816 conv1 output positions
#define N2_   (B_*F2_*HW2_)    // 100352 conv2 output positions
#define K2_   (C1_*27)         // 1728
#define KS2_  (K2_/32)         // 54 k-steps
#define NPOS2_PER_B (F2_*HW2_) // 25088

// ---------------- workspace layout (bytes, 256-aligned) ----------------
#define OFF_CODE 0ull                         // u8   [4*16*224*224]  = 3,211,264
#define OFF_FMAX (OFF_CODE + 3211264ull)      // int  [64]
#define OFF_LBP  (OFF_FMAX + 256ull)          // f16  lbp             = 6,422,528
#define OFF_ACT1 (OFF_LBP  + 6422528ull)      // f16  conv1 out [b][f][y][x][c64] = 102,760,448
#define OFF_WF1  (OFF_ACT1 + 102760448ull)    // f16  conv1 A-frags   = 4,096
#define OFF_WF2  (OFF_WF1  + 4096ull)         // f16  conv2 A-frags   = 442,368
#define OFF_ACC  (OFF_WF2  + 442368ull)       // f32  [512]
#define OFF_ZERO (OFF_ACC  + 2048ull)         // 64-byte zero page for OOB redirect

// ---------------- init: zero accumulators, frame maxima, zero page ----------------
__global__ void k_init(float* __restrict__ accum, int* __restrict__ fmax,
                       float* __restrict__ zp) {
  int t = threadIdx.x;
  if (t < 64) fmax[t] = 0;
  if (t < 16) zp[t] = 0.0f;
  accum[t] = 0.0f;
  accum[t + 256] = 0.0f;
}

// ---------------- uniform LBP codes (P=8, R=1), per-frame max ----------------
__global__ void k_lbp(const float* __restrict__ x, unsigned char* __restrict__ code,
                      int* __restrict__ fmax) {
  __shared__ float tile[18 * 18];
  __shared__ int smax;
  const int bf = blockIdx.z;               // 0..63  (b*16 + f)
  const int b = bf >> 4, f = bf & 15;
  const int ox = blockIdx.x * 16, oy = blockIdx.y * 16;
  const int tid = threadIdx.y * 16 + threadIdx.x;
  if (tid == 0) smax = 0;
  const float* x0 = x + ((size_t)(b * 3 + 0) * F_ + f) * HW_;
  const float* x1 = x + ((size_t)(b * 3 + 1) * F_ + f) * HW_;
  const float* x2 = x + ((size_t)(b * 3 + 2) * F_ + f) * HW_;
  for (int i = tid; i < 18 * 18; i += 256) {
    int r = i / 18, c = i % 18;
    int yy = oy + r - 1, xx = ox + c - 1;
    float g = 0.0f;
    if (yy >= 0 && yy < H_ && xx >= 0 && xx < W_) {
      int idx = yy * W_ + xx;
      g = floorf((x0[idx] + x1[idx] + x2[idx]) * (255.0f / 3.0f));
    }
    tile[i] = g;
  }
  __syncthreads();
  const int ly = threadIdx.y, lx = threadIdx.x;
#define T_(dr, dc) tile[(ly + 1 + (dr)) * 18 + (lx + 1 + (dc))]
  const float g = T_(0, 0);
  const int   R0[8] = { 0, -1, -1, -1,  0,  0,  1,  0};
  const int   C0[8] = { 1,  0,  0, -1, -1, -1,  0,  0};
  const float FR[8] = {0.f, 0.29289f, 0.f, 0.29289f, 0.f, 0.70711f, 0.f, 0.70711f};
  const float FC[8] = {0.f, 0.70711f, 0.f, 0.29289f, 0.f, 0.29289f, 0.f, 0.70711f};
  unsigned mask = 0u;
#pragma unroll
  for (int p = 0; p < 8; ++p) {
    float s;
    if (FR[p] == 0.f && FC[p] == 0.f) {
      s = T_(R0[p], C0[p]);
    } else {
      const float fr = FR[p], fc = FC[p];
      s = (1.f - fr) * (1.f - fc) * T_(R0[p],     C0[p])
        + (1.f - fr) * fc         * T_(R0[p],     C0[p] + 1)
        + fr         * (1.f - fc) * T_(R0[p] + 1, C0[p])
        + fr         * fc         * T_(R0[p] + 1, C0[p] + 1);
    }
    mask |= (s >= g) ? (1u << p) : 0u;
  }
#undef T_
  unsigned rot = ((mask << 1) | (mask >> 7)) & 0xffu;
  int changes = __popc(mask ^ rot);
  int lcode = (changes <= 2) ? __popc(mask) : 9;
  atomicMax(&smax, lcode);
  code[(size_t)bf * HW_ + (size_t)(oy + ly) * W_ + (ox + lx)] = (unsigned char)lcode;
  __syncthreads();
  if (tid == 0) atomicMax(&fmax[bf], smax);
}

// ---------------- normalize codes by per-frame max -> f16 ----------------
__global__ void k_norm(const unsigned char* __restrict__ code, const int* __restrict__ fmax,
                       _Float16* __restrict__ lbp) {
  size_t i = (size_t)blockIdx.x * 256 + threadIdx.x;   // 3,211,264 total
  int frame = (int)(i / HW_);
  int m = fmax[frame];
  float c = (float)code[i];
  float v = (m > 0) ? c / (float)m : c;
  lbp[i] = (_Float16)v;
}

// A-fragment (16x32 f16) inverse lane/elem -> k mapping (this code's contiguous order k = 16*(lane/16) + elem)
__device__ __forceinline__ int afrag_k(int lane, int elem) {
  int khi = lane >> 4;
  return elem + 16 * khi;
}

// ---------------- pack conv1 effective weights into A-fragments (K = tap, pad 27->32) ----------
__global__ void k_wprep1(const float* __restrict__ w1, _Float16* __restrict__ wf1) {
  int idx = blockIdx.x * 256 + threadIdx.x;   // 4 mt * 32 lane * 16 elem = 2048
  int mt = idx >> 9;
  int lane = (idx >> 4) & 31;
  int elem = idx & 15;
  int k = afrag_k(lane, elem);
  int mg = mt * 16 + (lane & 15);
  float v = 0.0f;
  if (k < 27)  // channels folded: lbp broadcast across 3 input channels
    v = w1[mg * 81 + k] + w1[mg * 81 + 27 + k] + w1[mg * 81 + 54 + k];
  wf1[idx] = (_Float16)v;
}

// ---------------- pack conv2 weights into A-fragments; K reordered as k' = tap*64 + cin ----------
__global__ void k_wprep2(const float* __restrict__ w2, _Float16* __restrict__ wf2) {
  int idx = blockIdx.x * 256 + threadIdx.x;   // 8 mt * 54 ks * 32 lane * 16 elem = 221184
  int mt = idx / (KS2_ * 512);
  int rem = idx % (KS2_ * 512);
  int ks = rem >> 9;
  int lane = (rem >> 4) & 31;
  int elem = idx & 15;
  int kp = ks * 32 + afrag_k(lane, elem);     // k' = r*64 + c
  int r = kp >> 6, c = kp & 63;
  int mg = mt * 16 + (lane & 15);
  wf2[idx] = (_Float16)w2[(size_t)mg * K2_ + c * 27 + r];
}

// ---------------- conv1 implicit GEMM: M=64, N-tile=32, K=32(pad 27); 8 waves = 4mt x 2sub -------
__global__ void __launch_bounds__(256) k_conv1(const _Float16* __restrict__ lbp,
                                               const _Float16* __restrict__ wf1,
                                               const float* __restrict__ b1,
                                               _Float16* __restrict__ act1) {
  __shared__ alignas(32) _Float16 sB[2][512];
  const int ntile = blockIdx.x;
  const int tid = threadIdx.x;
  for (int i = tid; i < 1024; i += 256) {     // gather two 32x16 im2col tiles
    int sub = i >> 9, j = i & 511;
    int kk = j >> 4, nn = j & 15;
    int n = ntile * 32 + sub * 16 + nn;
    int b = n / (F_ * HW1_), rem = n % (F_ * HW1_);
    int f = rem / HW1_, r2 = rem % HW1_;
    int y = r2 / W1_, xx = r2 % W1_;
    float v = 0.0f;
    if (kk < 27) {
      int kd = kk / 9, kh = (kk / 3) % 3, kw = kk % 3;
      int fi = f + kd - 1, yi = 2 * y + kh - 1, xi = 2 * xx + kw - 1;
      if (fi >= 0 && fi < F_ && yi >= 0 && yi < H_ && xi >= 0 && xi < W_)
        v = (float)lbp[((size_t)(b * F_ + fi) * H_ + yi) * W_ + xi];
    }
    int bl = (kk < 16) ? nn : nn + 16;        // B-frag: lanes 0-15 K=0..15, 16-31 K=16..31
    sB[sub][bl * 16 + (kk & 15)] = (_Float16)v;
  }
  __syncthreads();
  const int wave = tid >> 5, lane = tid & 31;
  const int mt = wave & 3, sub = wave >> 2;
  v16h a  = *(const v16h*)(wf1 + (size_t)(mt * 32 + lane) * 16);
  v16h bb = *(const v16h*)(&sB[sub][lane * 16]);
  v8f c = {};
  c = __builtin_amdgcn_wmma_f32_16x16x32_f16(false, a, false, bb, (short)0, c, false, false);
  // epilogue: bias + ReLU; channel-innermost layout [pos][c64] -> one b128 store per lane
  int n = ntile * 32 + sub * 16 + (lane & 15);
  int chbase = mt * 16 + 8 * (lane >> 4);     // C/D layout: VGPR e -> row e (+8 for hi lanes)
  v8h o;
#pragma unroll
  for (int e = 0; e < 8; ++e)
    o[e] = (_Float16)fmaxf(c[e] + b1[chbase + e], 0.0f);
  *(v8h*)(act1 + (size_t)n * C1_ + chbase) = o;
}

// ------- conv2 implicit GEMM: M=128 (8 waves), N-tile=32 (2 sub), K=1728; fused mean -------
// K order k' = tap*64 + cin: each 32-k step = one uniform tap + 32 contiguous channels.
// B tiles stream global->LDS with async dword copies (OOB -> zero page), double-buffered so
// the next tile is in flight (ASYNCcnt) while the current tile's WMMAs execute.
__global__ void __launch_bounds__(256) k_conv2(const _Float16* __restrict__ act1,
                                               const _Float16* __restrict__ wf2,
                                               const float* __restrict__ b2,
                                               const _Float16* __restrict__ zpage,
                                               float* __restrict__ accum) {
  __shared__ alignas(32) _Float16 sB[2][2][512];   // [buf][sub][frag]
  const int ntile = blockIdx.x;
  const int tid = threadIdx.x;
  const int wave = tid >> 5, lane = tid & 31;
  const _Float16* wf = wf2 + (size_t)wave * KS2_ * 512;
  // per-thread gather role: position nn within subtile, channel pair kk (dword)
  const int nn = tid >> 4;            // 0..15
  const int kk = (tid & 15) * 2;      // 0,2,..,30 within the 32-channel chunk
  const int bl = (kk < 16) ? nn : nn + 16;
  unsigned ldsOff[2][2];
#pragma unroll
  for (int bf = 0; bf < 2; ++bf)
#pragma unroll
    for (int s = 0; s < 2; ++s)
      ldsOff[bf][s] = (unsigned)(unsigned long long)&sB[bf][s][bl * 16 + (kk & 15)];
  // position decode (per subtile, loop-invariant)
  int posb[2], posf[2], posy[2], posx[2];
#pragma unroll
  for (int s = 0; s < 2; ++s) {
    int n = ntile * 32 + s * 16 + nn;
    posb[s] = n / NPOS2_PER_B;
    int rem = n % NPOS2_PER_B;
    posf[s] = rem / HW2_;
    int r2 = rem % HW2_;
    posy[s] = r2 / W2_;
    posx[s] = r2 % W2_;
  }
  // issue async copies of k-step ks's B tile into LDS buffer buf
  auto issueB = [&](int ks, int buf) {
    int r = ks >> 1;                            // uniform tap for whole k-step (SALU)
    int kd = r / 9, kh = (r / 3) % 3, kw = r % 3;
    int chunk = (ks & 1) * 32;                  // which 32-channel half
#pragma unroll
    for (int s = 0; s < 2; ++s) {
      int fi = 2 * posf[s] + kd - 1;
      int yi = 2 * posy[s] + kh - 1;
      int xi = 2 * posx[s] + kw - 1;
      bool ok = (fi >= 0 && fi < F_ && yi >= 0 && yi < H1_ && xi >= 0 && xi < W1_);
      const _Float16* src = ok
        ? act1 + ((size_t)((posb[s] * F_ + fi) * H1_ + yi) * W1_ + xi) * C1_ + chunk + kk
        : zpage + kk;
      unsigned long long ga = (unsigned long long)src;
      asm volatile("global_load_async_to_lds_b32 %0, %1, off"
                   :: "v"(ldsOff[buf][s]), "v"(ga) : "memory");
    }
  };
  v8f acc0 = {}, acc1 = {};
  issueB(0, 0);                                 // prologue: prime buffer 0
  for (int ks = 0; ks < KS2_; ++ks) {
    const int cur = ks & 1;
    // retire this wave's in-flight copies (they target buf cur), then rendezvous:
    // after the barrier every wave's copies for buf cur have landed in LDS, and every
    // wave has finished reading buf cur^1 (ds_load -> WMMA register dependency).
    asm volatile("s_wait_asynccnt 0x0" ::: "memory");
    __syncthreads();
    if (ks + 1 < KS2_) {
      issueB(ks + 1, cur ^ 1);                  // overlap next tile's copy with WMMAs
      __builtin_prefetch(wf + ((size_t)(ks + 1) * 32 + lane) * 16, 0, 1);
    }
    v16h a  = *(const v16h*)(wf + ((size_t)ks * 32 + lane) * 16);
    v16h b0 = *(const v16h*)(&sB[cur][0][lane * 16]);
    v16h b1v = *(const v16h*)(&sB[cur][1][lane * 16]);
    acc0 = __builtin_amdgcn_wmma_f32_16x16x32_f16(false, a, false, b0,  (short)0, acc0, false, false);
    acc1 = __builtin_amdgcn_wmma_f32_16x16x32_f16(false, a, false, b1v, (short)0, acc1, false, false);
  }
  // epilogue: bias + ReLU + partial global-mean reduction (32 columns, single batch b)
  int b = (ntile * 32) / NPOS2_PER_B;           // tiles never cross batch (25088 % 32 == 0)
  int chbase = wave * 16 + 8 * (lane >> 4);
  float vals[8];
#pragma unroll
  for (int e = 0; e < 8; ++e) {
    float bias = b2[chbase + e];
    vals[e] = fmaxf(acc0[e] + bias, 0.0f) + fmaxf(acc1[e] + bias, 0.0f);
  }
#pragma unroll
  for (int m = 1; m < 16; m <<= 1) {            // butterfly within 16-lane column groups
#pragma unroll
    for (int e = 0; e < 8; ++e)
      vals[e] += __shfl_xor(vals[e], m, 32);
  }
  if ((lane & 15) == 0) {
#pragma unroll
    for (int e = 0; e < 8; ++e)
      atomicAdd(&accum[b * C2_ + chbase + e], vals[e]);
  }
}

// ---------------- finalize: divide sums by F2*H2*W2 ----------------
__global__ void k_final(const float* __restrict__ accum, float* __restrict__ out) {
  int i = blockIdx.x * 256 + threadIdx.x;     // 512
  out[i] = accum[i] * (1.0f / (float)NPOS2_PER_B);
}

extern "C" void kernel_launch(void* const* d_in, const int* in_sizes, int n_in,
                              void* d_out, int out_size, void* d_ws, size_t ws_size,
                              hipStream_t stream) {
  const float* x  = (const float*)d_in[0];
  const float* w1 = (const float*)d_in[1];
  const float* b1 = (const float*)d_in[2];
  const float* w2 = (const float*)d_in[3];
  const float* b2 = (const float*)d_in[4];
  float* out = (float*)d_out;
  char* ws = (char*)d_ws;
  unsigned char* code = (unsigned char*)(ws + OFF_CODE);
  int* fmax      = (int*)(ws + OFF_FMAX);
  _Float16* lbp  = (_Float16*)(ws + OFF_LBP);
  _Float16* act1 = (_Float16*)(ws + OFF_ACT1);
  _Float16* wf1  = (_Float16*)(ws + OFF_WF1);
  _Float16* wf2  = (_Float16*)(ws + OFF_WF2);
  float* accum   = (float*)(ws + OFF_ACC);
  float* zpf     = (float*)(ws + OFF_ZERO);
  const _Float16* zpage = (const _Float16*)(ws + OFF_ZERO);

  k_init  <<<1, 256, 0, stream>>>(accum, fmax, zpf);
  k_lbp   <<<dim3(14, 14, 64), dim3(16, 16), 0, stream>>>(x, code, fmax);
  k_norm  <<<12544, 256, 0, stream>>>(code, fmax, lbp);
  k_wprep1<<<8, 256, 0, stream>>>(w1, wf1);
  k_wprep2<<<864, 256, 0, stream>>>(w2, wf2);
  k_conv1 <<<N1_ / 32, 256, 0, stream>>>(lbp, wf1, b1, act1);
  k_conv2 <<<N2_ / 32, 256, 0, stream>>>(act1, wf2, b2, zpage, accum);
  k_final <<<2, 256, 0, stream>>>(accum, out);
}
